// TrafficGATActor_81724637708833
// MI455X (gfx1250) — compile-verified, matrix operand order checked by reading, not run on hardware
//
#include <hip/hip_runtime.h>
#include <hip/hip_bf16.h>

// ---------- types for WMMA (CDNA5 / gfx1250, wave32) ----------
typedef __attribute__((ext_vector_type(16))) __bf16 v16bf;
typedef __attribute__((ext_vector_type(8)))  float  v8f;
typedef unsigned short u16;
typedef __attribute__((ext_vector_type(8)))  u16    u16x8;

union BF16Frag { u16x8 h[2]; v16bf v; };

__device__ __forceinline__ u16 f2bf(float f) {
    union { float f; unsigned u; } x; x.f = f;
    unsigned r = x.u + 0x7FFFu + ((x.u >> 16) & 1u);   // round-to-nearest-even
    return (u16)(r >> 16);
}

__device__ __forceinline__ float wred(float v) {
#pragma unroll
    for (int o = 16; o > 0; o >>= 1) v += __shfl_xor(v, o, 32);
    return v;
}

// float atomic max via int/uint ordering trick -> global_atomic_{max_i32,min_u32}
__device__ __forceinline__ void atomicMaxF(float* a, float v) {
    if (v >= 0.f) atomicMax((int*)a, __float_as_int(v));
    else          atomicMin((unsigned*)a, __float_as_uint(v));
}

__device__ __forceinline__ float gelu_exact(float v) {
    return 0.5f * v * (1.f + erff(v * 0.70710678118654752f));
}

// ---------- generic fill ----------
__global__ void k_fill(float* p, float v, long n) {
    long i = (long)blockIdx.x * blockDim.x + threadIdx.x;
    long st = (long)gridDim.x * blockDim.x;
    for (; i < n; i += st) p[i] = v;
}

// ---------- conversions ----------
__global__ void k_cvt_x(const float* __restrict__ x, u16* __restrict__ xb,
                        int N, int F, int Kpad, long total) {
    long i = (long)blockIdx.x * blockDim.x + threadIdx.x;
    if (i >= total) return;
    int n = (int)(i / Kpad), k = (int)(i % Kpad);
    xb[i] = (n < N && k < F) ? f2bf(x[(size_t)n * F + k]) : (u16)0;
}

// Wt[n*Kpad + k] = bf16(W[k*Nout + n])  (transpose + pad K)
__global__ void k_cvt_wt(const float* __restrict__ W, u16* __restrict__ Wt,
                         int K, int Nout, int Kpad, long total) {
    long i = (long)blockIdx.x * blockDim.x + threadIdx.x;
    if (i >= total) return;
    int n = (int)(i / Kpad), k = (int)(i % Kpad);
    Wt[i] = (k < K) ? f2bf(W[(size_t)k * Nout + n]) : (u16)0;
}

// ---------- bf16 WMMA GEMM: C[M,Nout] = A[M,Kpad] * Bt[Nout,Kpad]^T ----------
// One wave computes one 16x16 output tile; fragments loaded straight from
// global as 2x16B per lane matching the ISA 16-bit A/B lane layout.
__global__ void k_gemm_bf16(const u16* __restrict__ A, const u16* __restrict__ Bt,
                            float* __restrict__ C, int Mtiles, int Ntiles,
                            int Kpad, int ldc, int Mrows) {
    int tile = blockIdx.x * (blockDim.x >> 5) + (threadIdx.x >> 5);
    if (tile >= Mtiles * Ntiles) return;
    int tm = tile / Ntiles, tn = tile - tm * Ntiles;
    int lane = threadIdx.x & 31;
    int sub  = lane & 15;
    int kb   = (lane >> 4) << 3;  // 0 or 8
    const u16* arow = A  + (size_t)(tm * 16 + sub) * Kpad + kb;
    const u16* brow = Bt + (size_t)(tn * 16 + sub) * Kpad + kb;
    v8f acc = {0.f, 0.f, 0.f, 0.f, 0.f, 0.f, 0.f, 0.f};
    for (int k0 = 0; k0 < Kpad; k0 += 32) {
        BF16Frag fa, fb;
        fa.h[0] = *(const u16x8*)(arow + k0);
        fa.h[1] = *(const u16x8*)(arow + k0 + 16);
        fb.h[0] = *(const u16x8*)(brow + k0);
        fb.h[1] = *(const u16x8*)(brow + k0 + 16);
        acc = __builtin_amdgcn_wmma_f32_16x16x32_bf16(
                  false, fa.v, false, fb.v, (short)0, acc, false, false);
    }
    int r0  = tm * 16 + ((lane >> 4) << 3);
    int col = tn * 16 + sub;
#pragma unroll
    for (int g = 0; g < 8; ++g) {
        int r = r0 + g;
        if (r < Mrows) C[(size_t)r * ldc + col] = acc[g];
    }
}

// ---------- self-loop edge attr (mean of incoming) ----------
__global__ void k_loop_stats(const int* __restrict__ ei, const float* __restrict__ ea,
                             float* cnt, float* asum, int E) {
    int e = blockIdx.x * blockDim.x + threadIdx.x;
    if (e >= E) return;
    int d = ei[E + e];
    atomicAdd(&cnt[d], 1.f);
#pragma unroll
    for (int j = 0; j < 4; ++j) atomicAdd(&asum[d * 4 + j], ea[(size_t)e * 4 + j]);
}

__global__ void k_loop_fin(const float* cnt, const float* asum, float* la, int N) {
    int n = blockIdx.x * blockDim.x + threadIdx.x;
    if (n >= N) return;
    float c = fmaxf(cnt[n], 1.f);
#pragma unroll
    for (int j = 0; j < 4; ++j) la[n * 4 + j] = asum[n * 4 + j] / c;
}

// ---------- fold (We @ a_edge) -> q[E_DIM, H] ----------
__global__ void k_qvec(const float* We1, const float* ae1,
                       const float* We2, const float* ae2,
                       float* q1, float* q2) {
    int t = threadIdx.x;
    if (t < 16) {
        int j = t >> 2, h = t & 3;
        float s = 0.f;
        for (int d = 0; d < 128; ++d) s += We1[j * 512 + h * 128 + d] * ae1[h * 128 + d];
        q1[j * 4 + h] = s;
    } else if (t < 20) {
        int j = t - 16;
        float s = 0.f;
        for (int d = 0; d < 128; ++d) s += We2[j * 128 + d] * ae2[d];
        q2[j] = s;
    }
}

// ---------- per-node attention dots: s_src/s_dst ----------
__global__ void k_sdots(const float* __restrict__ xs, const float* __restrict__ a_s,
                        const float* __restrict__ a_d, float* os, float* od,
                        int NH, int H, int D) {
    int u = blockIdx.x * (blockDim.x >> 5) + (threadIdx.x >> 5);
    if (u >= NH) return;
    int lane = threadIdx.x & 31;
    int n = u / H, h = u - n * H;
    const float4 xv = *(const float4*)&xs[(size_t)n * H * D + h * D + lane * 4];
    const float4 sv = *(const float4*)&a_s[h * D + lane * 4];
    const float4 dv = *(const float4*)&a_d[h * D + lane * 4];
    float ps = xv.x * sv.x + xv.y * sv.y + xv.z * sv.z + xv.w * sv.w;
    float pd = xv.x * dv.x + xv.y * dv.y + xv.z * dv.z + xv.w * dv.w;
    ps = wred(ps); pd = wred(pd);
    if (lane == 0) { os[u] = ps; od[u] = pd; }
}

// ---------- edge pass A: leaky-relu score + segment max ----------
__global__ void k_edge_scores(const int* __restrict__ ei, const float* __restrict__ ea,
                              const float* __restrict__ la, const float* __restrict__ ss,
                              const float* __restrict__ sd, const float* __restrict__ q,
                              float* alpha, float* mbuf, int E, int Etot, int H) {
    int id = blockIdx.x * blockDim.x + threadIdx.x;
    if (id >= Etot * H) return;
    int e = id / H, h = id - e * H;
    int s, d; const float* a;
    if (e < E) { s = ei[e]; d = ei[E + e]; a = ea + (size_t)e * 4; }
    else       { s = d = e - E;            a = la + (size_t)s * 4; }
    float se = a[0] * q[h] + a[1] * q[H + h] + a[2] * q[2 * H + h] + a[3] * q[3 * H + h];
    float al = ss[s * H + h] + sd[d * H + h] + se;
    al = (al > 0.f) ? al : 0.2f * al;
    alpha[id] = al;
    atomicMaxF(&mbuf[d * H + h], al);
}

// ---------- edge pass B: exp + segment sum ----------
__global__ void k_edge_exp(const int* __restrict__ ei, const float* __restrict__ mbuf,
                           float* alpha, float* den, int E, int Etot, int H) {
    int id = blockIdx.x * blockDim.x + threadIdx.x;
    if (id >= Etot * H) return;
    int e = id / H, h = id - e * H;
    int d = (e < E) ? ei[E + e] : (e - E);
    float ex = expf(alpha[id] - mbuf[d * H + h]);
    alpha[id] = ex;
    atomicAdd(&den[d * H + h], ex);
}

// ---------- edge pass C, layer1: 4 heads x 128, one wave per edge ----------
__global__ void k_msg1(const int* __restrict__ ei, const float* __restrict__ alpha,
                       const float* __restrict__ den, const float* __restrict__ xs,
                       float* __restrict__ agg, int E, int Etot) {
    int e = blockIdx.x * (blockDim.x >> 5) + (threadIdx.x >> 5);
    if (e >= Etot) return;
    int lane = threadIdx.x & 31;
    int s, d;
    if (e < E) { s = ei[e]; d = ei[E + e]; } else { s = d = e - E; }
    float4 av = *(const float4*)&alpha[(size_t)e * 4];
    float4 dv = *(const float4*)&den[(size_t)d * 4];
    float w[4] = { av.x / (dv.x + 1e-16f), av.y / (dv.y + 1e-16f),
                   av.z / (dv.z + 1e-16f), av.w / (dv.w + 1e-16f) };
    const float* xr = xs + (size_t)s * 512;
    float* ar = agg + (size_t)d * 512;
#pragma unroll
    for (int i = 0; i < 4; ++i) {
        int c0 = i * 128 + lane * 4;
        float4 xv = *(const float4*)&xr[c0];
        atomicAdd(&ar[c0 + 0], xv.x * w[i]);
        atomicAdd(&ar[c0 + 1], xv.y * w[i]);
        atomicAdd(&ar[c0 + 2], xv.z * w[i]);
        atomicAdd(&ar[c0 + 3], xv.w * w[i]);
    }
}

// ---------- edge pass C, layer2: 1 head x 128 ----------
__global__ void k_msg2(const int* __restrict__ ei, const float* __restrict__ alpha,
                       const float* __restrict__ den, const float* __restrict__ xs,
                       float* __restrict__ agg, int E, int Etot) {
    int e = blockIdx.x * (blockDim.x >> 5) + (threadIdx.x >> 5);
    if (e >= Etot) return;
    int lane = threadIdx.x & 31;
    int s, d;
    if (e < E) { s = ei[e]; d = ei[E + e]; } else { s = d = e - E; }
    float w = alpha[e] / (den[d] + 1e-16f);
    int c0 = lane * 4;
    float4 xv = *(const float4*)&xs[(size_t)s * 128 + c0];
    float* ar = agg + (size_t)d * 128;
    atomicAdd(&ar[c0 + 0], xv.x * w);
    atomicAdd(&ar[c0 + 1], xv.y * w);
    atomicAdd(&ar[c0 + 2], xv.z * w);
    atomicAdd(&ar[c0 + 3], xv.w * w);
}

// ---------- bias + GELU + bf16 pack (feeds GEMM2) ----------
__global__ void k_gelu_bf(const float* __restrict__ agg, const float* __restrict__ b,
                          u16* __restrict__ xb, int N, int Mpad) {
    long i = (long)blockIdx.x * blockDim.x + threadIdx.x;
    long total = (long)Mpad * 512;
    if (i >= total) return;
    int n = (int)(i >> 9);
    int c = (int)(i & 511);
    u16 o = 0;
    if (n < N) o = f2bf(gelu_exact(agg[i] + b[c]));
    xb[i] = o;
}

// ---------- bias + GELU + LayerNorm + two head GEMMs, one wave per node ----------
__global__ void k_final(const float* __restrict__ agg2, const float* __restrict__ b2,
                        const float* __restrict__ g,   const float* __restrict__ bb,
                        const float* __restrict__ pw,  const float* __restrict__ pb,
                        const float* __restrict__ dwt, const float* __restrict__ dbb,
                        float* __restrict__ out, int N) {
    int n = blockIdx.x * (blockDim.x >> 5) + (threadIdx.x >> 5);
    if (n >= N) return;
    int lane = threadIdx.x & 31;
    int c0 = lane * 4;
    float4 v = *(const float4*)&agg2[(size_t)n * 128 + c0];
    float x0 = gelu_exact(v.x + b2[c0 + 0]);
    float x1 = gelu_exact(v.y + b2[c0 + 1]);
    float x2 = gelu_exact(v.z + b2[c0 + 2]);
    float x3 = gelu_exact(v.w + b2[c0 + 3]);
    float mu = wred(x0 + x1 + x2 + x3) * (1.f / 128.f);
    float d0 = x0 - mu, d1 = x1 - mu, d2 = x2 - mu, d3 = x3 - mu;
    float var = wred(d0 * d0 + d1 * d1 + d2 * d2 + d3 * d3) * (1.f / 128.f);
    float rn = rsqrtf(var + 1e-5f);
    float y0 = d0 * rn * g[c0 + 0] + bb[c0 + 0];
    float y1 = d1 * rn * g[c0 + 1] + bb[c0 + 1];
    float y2 = d2 * rn * g[c0 + 2] + bb[c0 + 2];
    float y3 = d3 * rn * g[c0 + 3] + bb[c0 + 3];
    float* o1 = out + (size_t)n * 8;
    float* o2 = out + (size_t)N * 8 + (size_t)n * 4;
#pragma unroll
    for (int j = 0; j < 8; ++j) {
        float p = y0 * pw[(c0 + 0) * 8 + j] + y1 * pw[(c0 + 1) * 8 + j]
                + y2 * pw[(c0 + 2) * 8 + j] + y3 * pw[(c0 + 3) * 8 + j];
        p = wred(p);
        if (lane == 0) o1[j] = p + pb[j];
    }
#pragma unroll
    for (int j = 0; j < 4; ++j) {
        float p = y0 * dwt[(c0 + 0) * 4 + j] + y1 * dwt[(c0 + 1) * 4 + j]
                + y2 * dwt[(c0 + 2) * 4 + j] + y3 * dwt[(c0 + 3) * 4 + j];
        p = wred(p);
        if (lane == 0) o2[j] = p + dbb[j];
    }
}

extern "C" void kernel_launch(void* const* d_in, const int* in_sizes, int n_in,
                              void* d_out, int out_size, void* d_ws, size_t ws_size,
                              hipStream_t stream) {
    const float* x   = (const float*)d_in[0];
    const int*   ei  = (const int*)  d_in[1];
    const float* ea  = (const float*)d_in[2];
    const float* W1  = (const float*)d_in[3];
    const float* as1 = (const float*)d_in[4];
    const float* ad1 = (const float*)d_in[5];
    const float* We1 = (const float*)d_in[6];
    const float* ae1 = (const float*)d_in[7];
    const float* b1  = (const float*)d_in[8];
    const float* W2  = (const float*)d_in[9];
    const float* as2 = (const float*)d_in[10];
    const float* ad2 = (const float*)d_in[11];
    const float* We2 = (const float*)d_in[12];
    const float* ae2 = (const float*)d_in[13];
    const float* b2  = (const float*)d_in[14];
    const float* lng = (const float*)d_in[15];
    const float* lnb = (const float*)d_in[16];
    const float* pw  = (const float*)d_in[17];
    const float* pb  = (const float*)d_in[18];
    const float* dwt = (const float*)d_in[19];
    const float* dbb = (const float*)d_in[20];
    float* out = (float*)d_out;
    (void)n_in; (void)out_size; (void)ws_size;

    const int N    = in_sizes[0] / 70;
    const int E    = in_sizes[1] / 2;
    const int Etot = E + N;
    const int Mpad = ((N + 15) / 16) * 16;
    const int H = 4, KP1 = 96, C1 = 512, KP2 = 512, C2 = 128;

    // bump allocator over d_ws
    size_t off = 0;
    auto alloc = [&](size_t b) -> void* {
        off = (off + 255) & ~(size_t)255;
        void* p = (char*)d_ws + off;
        off += b;
        return p;
    };
    u16*   Xbf  = (u16*)  alloc((size_t)Mpad * KP1 * 2);
    u16*   W1t  = (u16*)  alloc((size_t)C1 * KP1 * 2);
    u16*   W2t  = (u16*)  alloc((size_t)C2 * KP2 * 2);
    float* q1   = (float*)alloc(16 * 4);
    float* q2   = (float*)alloc(4 * 4);
    float* cnt  = (float*)alloc((size_t)N * 4);
    float* asum = (float*)alloc((size_t)N * 4 * 4);
    float* la   = (float*)alloc((size_t)N * 4 * 4);
    float* ss1  = (float*)alloc((size_t)N * H * 4);
    float* sd1  = (float*)alloc((size_t)N * H * 4);
    float* m1   = (float*)alloc((size_t)N * H * 4);
    float* dn1  = (float*)alloc((size_t)N * H * 4);
    float* al1  = (float*)alloc((size_t)Etot * H * 4);
    float* R1   = (float*)alloc((size_t)Mpad * C1 * 4);   // xs1 -> later xs2 + x2bf
    float* R2   = (float*)alloc((size_t)Mpad * C1 * 4);   // agg1 -> later agg2
    float* ss2  = (float*)alloc((size_t)N * 4);
    float* sd2  = (float*)alloc((size_t)N * 4);
    float* m2   = (float*)alloc((size_t)N * 4);
    float* dn2  = (float*)alloc((size_t)N * 4);
    float* al2  = (float*)alloc((size_t)Etot * 4);

    float* xs1  = R1;
    float* agg1 = R2;
    float* xs2  = R1;                                        // xs1 dead by then
    u16*   x2bf = (u16*)((char*)R1 + (size_t)Mpad * C2 * 4); // disjoint from xs2
    float* agg2 = R2;                                        // agg1 dead by then

    const int T = 256;
    auto nb = [&](long n) { return (unsigned)((n + T - 1) / T); };

    // conversions + q folding
    k_cvt_x <<<nb((long)Mpad * KP1), T, 0, stream>>>(x, Xbf, N, 70, KP1, (long)Mpad * KP1);
    k_cvt_wt<<<nb((long)C1 * KP1),   T, 0, stream>>>(W1, W1t, 70, C1, KP1, (long)C1 * KP1);
    k_cvt_wt<<<nb((long)C2 * KP2),   T, 0, stream>>>(W2, W2t, 512, C2, KP2, (long)C2 * KP2);
    k_qvec  <<<1, 32, 0, stream>>>(We1, ae1, We2, ae2, q1, q2);

    // self-loop attrs
    k_fill<<<nb(N), T, 0, stream>>>(cnt, 0.f, N);
    k_fill<<<nb((long)N * 4), T, 0, stream>>>(asum, 0.f, (long)N * 4);
    k_loop_stats<<<nb(E), T, 0, stream>>>(ei, ea, cnt, asum, E);
    k_loop_fin  <<<nb(N), T, 0, stream>>>(cnt, asum, la, N);

    // GEMM1 via WMMA: xs1 = X @ W1
    {
        long tiles = (long)(Mpad / 16) * (C1 / 16);
        k_gemm_bf16<<<nb(tiles * 32), T, 0, stream>>>(Xbf, W1t, xs1, Mpad / 16, C1 / 16, KP1, C1, N);
    }
    // layer1 attention + aggregation
    k_sdots<<<nb((long)N * H * 32), T, 0, stream>>>(xs1, as1, ad1, ss1, sd1, N * H, H, 128);
    k_fill<<<nb((long)N * H), T, 0, stream>>>(m1, -3.0e38f, (long)N * H);
    k_fill<<<nb((long)N * H), T, 0, stream>>>(dn1, 0.f, (long)N * H);
    k_fill<<<nb((long)Mpad * C1), T, 0, stream>>>(agg1, 0.f, (long)Mpad * C1);
    k_edge_scores<<<nb((long)Etot * H), T, 0, stream>>>(ei, ea, la, ss1, sd1, q1, al1, m1, E, Etot, H);
    k_edge_exp   <<<nb((long)Etot * H), T, 0, stream>>>(ei, m1, al1, dn1, E, Etot, H);
    k_msg1       <<<nb((long)Etot * 32), T, 0, stream>>>(ei, al1, dn1, xs1, agg1, E, Etot);

    // bias + GELU -> bf16 activations for GEMM2
    k_gelu_bf<<<nb((long)Mpad * C1), T, 0, stream>>>(agg1, b1, x2bf, N, Mpad);

    // GEMM2 via WMMA: xs2 = gelu(x) @ W2
    {
        long tiles = (long)(Mpad / 16) * (C2 / 16);
        k_gemm_bf16<<<nb(tiles * 32), T, 0, stream>>>(x2bf, W2t, xs2, Mpad / 16, C2 / 16, KP2, C2, N);
    }
    // layer2 attention + aggregation
    k_sdots<<<nb((long)N * 32), T, 0, stream>>>(xs2, as2, ad2, ss2, sd2, N, 1, 128);
    k_fill<<<nb(N), T, 0, stream>>>(m2, -3.0e38f, N);
    k_fill<<<nb(N), T, 0, stream>>>(dn2, 0.f, N);
    k_fill<<<nb((long)Mpad * C2), T, 0, stream>>>(agg2, 0.f, (long)Mpad * C2);
    k_edge_scores<<<nb((long)Etot), T, 0, stream>>>(ei, ea, la, ss2, sd2, q2, al2, m2, E, Etot, 1);
    k_edge_exp   <<<nb((long)Etot), T, 0, stream>>>(ei, m2, al2, dn2, E, Etot, 1);
    k_msg2       <<<nb((long)Etot * 32), T, 0, stream>>>(ei, al2, dn2, xs2, agg2, E, Etot);

    // bias + GELU + LayerNorm + policy/value heads
    k_final<<<nb((long)N * 32), T, 0, stream>>>(agg2, b2, lng, lnb, pw, pb, dwt, dbb, out, N);
}
